// MatchLSTM_69947837382660
// MI455X (gfx1250) — matrix operand order repaired and checked by reading, not compiled, not run on hardware
//
#include <hip/hip_runtime.h>
#include <hip/hip_bf16.h>

// ---------------------------------------------------------------------------
// Match-LSTM for MI455X (gfx1250, wave32, WMMA + TDM).
// Sizes: B=32, D=512, Q=64, H2=512.  K3 = d+ctx+h = 1536, NG = 4*H2 = 2048.
// ---------------------------------------------------------------------------

typedef unsigned short u16;
typedef unsigned int   u32;
typedef __attribute__((ext_vector_type(16))) __bf16 v16bf;
typedef __attribute__((ext_vector_type(8)))  float  v8f;
typedef __attribute__((ext_vector_type(4)))  unsigned int u32x4;
typedef __attribute__((ext_vector_type(8)))  int          i32x8;
typedef __attribute__((ext_vector_type(4)))  int          i32x4;

union ABf { v16bf v; uint4 q[2]; };

#define BB   32
#define DD   512
#define QQ   64
#define HH2  512
#define KK3  1536
#define NNG  2048

#if defined(__has_builtin)
#if __has_builtin(__builtin_amdgcn_tensor_load_to_lds) && __has_builtin(__builtin_amdgcn_s_wait_tensorcnt)
#define HAVE_TDM 1
#endif
#if __has_builtin(__builtin_amdgcn_tensor_store_from_lds) && __has_builtin(__builtin_amdgcn_s_wait_tensorcnt)
#define HAVE_TDM_ST 1
#endif
#endif

__device__ __forceinline__ u16 f2bf(float f) {
    u32 u = __builtin_bit_cast(u32, f);
    u32 r = (u + 0x7FFFu + ((u >> 16) & 1u)) >> 16;
    return (u16)r;
}
__device__ __forceinline__ float bf2f(u16 h) {
    u32 u = ((u32)h) << 16;
    return __builtin_bit_cast(float, u);
}
__device__ __forceinline__ float sigf(float x) { return 1.0f / (1.0f + __expf(-x)); }

// Fragment loads: hoisted per-lane base pointer + (mostly constant) element
// offset, so tile selection folds into the instruction's immediate field.
// A lane base: plane + (lane&15)*ld + ((lane>>4)<<4)
__device__ __forceinline__ v16bf loadA_off(const u16* lanebase, int koff) {
    const uint4* p0 = (const uint4*)(lanebase + koff);
    const uint4* p1 = (const uint4*)(lanebase + koff + 8);
    ABf a; a.q[0] = *p0; a.q[1] = *p1;
    return a.v;
}
// B lane base: WT + (n0 + (lane&15))*K + ((lane>>4)<<4)
__device__ __forceinline__ v16bf loadB_off(const u16* lanebase, int eoff) {
    const uint4* p = (const uint4*)(lanebase + eoff);
    ABf b; b.q[0] = p[0]; b.q[1] = p[1];
    return b.v;
}

// TDM 2D tile load: 16 rows x 512 bf16 elems, row stride rs elems -> LDS.
__device__ __forceinline__ void tdm_load_tile(const u16* gsrc, u16* ldsdst, u32 rs) {
#if defined(HAVE_TDM)
    unsigned long long ga = (unsigned long long)gsrc;
    u32x4 g0;
    g0[0] = 1u;                                           // count=1, user mode
    g0[1] = (u32)(unsigned long long)ldsdst;              // lds_addr (bytes)
    g0[2] = (u32)ga;                                      // global_addr[31:0]
    g0[3] = ((u32)(ga >> 32) & 0x01FFFFFFu) | (2u << 30); // addr[56:32] | type=2
    i32x8 g1;
    g1[0] = 0x10000;                  // workgroup_mask=0, data_size=1 (2B)
    g1[1] = (int)(512u << 16);        // tensor_dim0 = 512
    g1[2] = (int)(16u  << 16);        // tensor_dim1 = 16
    g1[3] = (int)(512u << 16);        // tile_dim0 = 512
    g1[4] = 16;                       // tile_dim1 = 16, tile_dim2 = 0
    g1[5] = (int)rs;                  // tensor_dim0_stride (elems)
    g1[6] = 0;
    g1[7] = 0;
    i32x4 gz; gz[0] = 0; gz[1] = 0; gz[2] = 0; gz[3] = 0;
#if __clang_major__ >= 23
    i32x8 g5; for (int i = 0; i < 8; ++i) g5[i] = 0;
    __builtin_amdgcn_tensor_load_to_lds(g0, g1, gz, gz, g5, 0);
#else
    __builtin_amdgcn_tensor_load_to_lds(g0, g1, gz, gz, 0);
#endif
#endif
}

// TDM 2D tile store: LDS -> 16 rows x 512 bf16 elems, row stride rs elems.
__device__ __forceinline__ void tdm_store_tile(u16* gdst, const u16* ldssrc, u32 rs) {
#if defined(HAVE_TDM_ST)
    unsigned long long ga = (unsigned long long)gdst;
    u32x4 g0;
    g0[0] = 1u;
    g0[1] = (u32)(unsigned long long)ldssrc;
    g0[2] = (u32)ga;
    g0[3] = ((u32)(ga >> 32) & 0x01FFFFFFu) | (2u << 30);
    i32x8 g1;
    g1[0] = 0x10000;
    g1[1] = (int)(512u << 16);
    g1[2] = (int)(16u  << 16);
    g1[3] = (int)(512u << 16);
    g1[4] = 16;
    g1[5] = (int)rs;
    g1[6] = 0;
    g1[7] = 0;
    i32x4 gz; gz[0] = 0; gz[1] = 0; gz[2] = 0; gz[3] = 0;
#if __clang_major__ >= 23
    i32x8 g5; for (int i = 0; i < 8; ++i) g5[i] = 0;
    __builtin_amdgcn_tensor_store_from_lds(g0, g1, gz, gz, g5, 0);
#else
    __builtin_amdgcn_tensor_store_from_lds(g0, g1, gz, gz, 0);
#endif
#endif
}

// ------------------------- one-time prep kernels ---------------------------

__global__ __launch_bounds__(256) void cvt_bf16_kernel(const float* __restrict__ in,
                                                       u16* __restrict__ out, int n) {
    for (int i = blockIdx.x * blockDim.x + threadIdx.x; i < n; i += gridDim.x * blockDim.x)
        out[i] = f2bf(in[i]);
}

// W [K,N] f32 row-major  ->  WT [N,K] bf16 (coalesced writes)
__global__ __launch_bounds__(256) void transpose_bf16_kernel(const float* __restrict__ W,
                                                             u16* __restrict__ WT,
                                                             int K, int N) {
    int total = K * N;
    for (int i = blockIdx.x * blockDim.x + threadIdx.x; i < total; i += gridDim.x * blockDim.x) {
        int nn = i / K, kk = i % K;
        WT[i] = f2bf(W[(size_t)kk * N + nn]);
    }
}

// Fused gate weight, transposed: WgT [2048, 1536]; K rows = [W_ih(1024) ; W_hh(512)]
__global__ __launch_bounds__(256) void build_wgt_kernel(const float* __restrict__ Wih,
                                                        const float* __restrict__ Whh,
                                                        u16* __restrict__ WgT) {
    int total = NNG * KK3;
    for (int i = blockIdx.x * blockDim.x + threadIdx.x; i < total; i += gridDim.x * blockDim.x) {
        int nn = i / KK3, kk = i % KK3;
        float v = (kk < 1024) ? Wih[(size_t)kk * NNG + nn]
                              : Whh[(size_t)(kk - 1024) * NNG + nn];
        WgT[i] = f2bf(v);
    }
}

__global__ __launch_bounds__(256) void qmean_kernel(const float* __restrict__ qry,
                                                    const float* __restrict__ qm,
                                                    float* __restrict__ qmean) {
    int tid = blockIdx.x * blockDim.x + threadIdx.x;
    if (tid >= BB * HH2) return;
    int b = tid >> 9, h = tid & 511;
    float s = 0.f, dn = 0.f;
    for (int q = 0; q < QQ; ++q) {
        float m = qm[b * QQ + q];
        s += qry[((size_t)(b * QQ + q)) * HH2 + h] * m;
        dn += m;
    }
    qmean[tid] = s / fmaxf(dn, 1.f);
}

__global__ __launch_bounds__(256) void h0_kernel(const float* __restrict__ qmean,
                                                 const float* __restrict__ Winit,
                                                 float* __restrict__ h0) {
    int tid = blockIdx.x * blockDim.x + threadIdx.x;
    if (tid >= BB * HH2) return;
    int b = tid >> 9, n = tid & 511;
    float acc = 0.f;
    for (int k = 0; k < HH2; ++k)
        acc += qmean[b * HH2 + k] * Winit[(size_t)k * HH2 + n];
    h0[tid] = tanhf(acc);
}

// ---------------------- generic bf16 WMMA GEMM (TN) ------------------------
// C[M,N] (f32) = A[M,K](bf16 row-major) x BT[N,K](bf16) + bias[N].
template <int K>
__global__ __launch_bounds__(256) void gemm_bf16_tn(const u16* __restrict__ A,
                                                    const u16* __restrict__ BT,
                                                    const float* __restrict__ bias,
                                                    float* __restrict__ C,
                                                    int M, int N) {
    __shared__ u16 Ash[16 * K];
    int tid = threadIdx.x;
    size_t mrow = (size_t)blockIdx.x * 16;

    constexpr int qperrow = K >> 3;
    for (int i = tid; i < 16 * qperrow; i += 256) {
        int row = i / qperrow, qi = i % qperrow;
        ((uint4*)(Ash + row * K))[qi] = ((const uint4*)(A + (mrow + row) * (size_t)K))[qi];
    }
    __syncthreads();

    int wv = tid >> 5, lane = tid & 31;
    int n = lane & 15, mhalf = (lane >> 4) << 3;
    int nbase = blockIdx.y * 512 + wv * 64;
    if (nbase >= N) return;               // wave-uniform

    v8f acc[4];
    for (int j = 0; j < 4; ++j) {
        float bv = bias ? bias[nbase + j * 16 + n] : 0.f;
        v8f t;
        for (int r = 0; r < 8; ++r) t[r] = bv;
        acc[j] = t;
    }

    const u16* ap  = Ash + (lane & 15) * K + ((lane >> 4) << 4);
    const u16* bp0 = BT + (size_t)(nbase + n) * K + ((lane >> 4) << 4);  // single base

    constexpr int ks = K >> 5;
#pragma unroll 4
    for (int kk = 0; kk < ks; ++kk) {
        const u16* bk = bp0 + (size_t)kk * 32;
        v16bf af = loadA_off(ap, kk * 32);
#pragma unroll
        for (int j = 0; j < 4; ++j) {
            if ((kk & 3) == 0) __builtin_prefetch(bk + j * 16 * K + 8 * 32, 0, 1);
            v16bf bfr = loadB_off(bk, j * 16 * K);   // tile offset folds to imm
            acc[j] = __builtin_amdgcn_wmma_f32_16x16x32_bf16(
                false, af, false, bfr, (short)0, acc[j], false, false);
        }
    }
    for (int j = 0; j < 4; ++j)
        for (int r = 0; r < 8; ++r)
            C[(mrow + mhalf + r) * (size_t)N + nbase + j * 16 + n] = acc[j][r];
}

// ------------------------- the recurrent kernel ----------------------------
// grid.x = 4 : dir = blockIdx.x>>1, batch tile = blockIdx.x&1 (16 rows each).
// 512 threads = 16 waves; wave w owns H-columns [w*32, w*32+32).
__global__ __launch_bounds__(512) void matchlstm_rnn_kernel(
        const u16*  __restrict__ docbf,   // [B,D,512] bf16
        const float* __restrict__ docWd,  // [B,D,512] f32 (includes b_g)
        const float* __restrict__ qWq,    // [B,Q,512] f32
        const u16*  __restrict__ qrybf,   // [B,Q,512] bf16
        const float* __restrict__ qm,     // [B,Q]
        const float* __restrict__ w_s,    // [512]
        const float* __restrict__ b_lstm, // [2,2048]
        const u16*  __restrict__ WrT,     // [512,512] bf16 (N,K)
        const u16*  __restrict__ WgT,     // [2][2048,1536] bf16 (N,K)
        const float* __restrict__ h0,     // [B,512]
        u16*        __restrict__ hrbf)    // [B,D,1024] bf16
{
    extern __shared__ char smem[];
    u16*   dsh   = (u16*)smem;                       // 16384 B
    u16*   ctxsh = (u16*)(smem + 16384);             // 16384 B
    u16*   hsh   = (u16*)(smem + 32768);             // 16384 B
    float* csh   = (float*)(smem + 49152);           // 32768 B
    float* presh = (float*)(smem + 81920);           // 32768 B
    float* scsh  = (float*)(smem + 114688);          //  4096 B
    float* wssh  = (float*)(smem + 118784);          //  2048 B

    const int tid    = threadIdx.x;
    const int dir    = blockIdx.x >> 1;
    const int bgbase = (blockIdx.x & 1) * 16;
    const u16* WG    = WgT + (size_t)dir * NNG * KK3;

    if (tid < HH2) wssh[tid] = w_s[tid];
    for (int i = tid; i < 16 * HH2; i += 512) {
        int b = i >> 9, h = i & 511;
        hsh[b * HH2 + h] = f2bf(h0[(bgbase + b) * HH2 + h]);
        csh[i] = 0.f;
    }
    __syncthreads();

    const int wv = tid >> 5, lane = tid & 31;
    const int n = lane & 15, mhalf = (lane >> 4) << 3;
    const int hc0 = wv * 32;

    // single hoisted per-lane base pointers; all tile selection via constant
    // element offsets that fold into the 24-bit instruction offset field.
    const u16* aph  = hsh + (lane & 15) * HH2 + ((lane >> 4) << 4);
    const u16* bpr0 = WrT + (size_t)(hc0 + n) * HH2 + ((lane >> 4) << 4);
    const u16* bpg0 = WG + (size_t)(hc0 + n) * KK3 + ((lane >> 4) << 4);

    for (int t = 0; t < DD; ++t) {
        const int tt = dir ? (DD - 1 - t) : t;

        // ---- phase 0: d_i tile -> LDS via TDM DMA (hidden behind phases 1-3)
#if defined(HAVE_TDM)
        if (tid < 32)
            tdm_load_tile(docbf + (((size_t)bgbase * DD + tt) << 9), dsh,
                          (u32)DD * 512u);
#else
        for (int i = tid; i < 16 * 64; i += 512) {
            int row = i >> 6, qi = i & 63;
            const uint4* src =
                (const uint4*)(docbf + (((size_t)(bgbase + row) * DD + tt) << 9)) + qi;
            ((uint4*)(dsh + row * HH2))[qi] = *src;
        }
#endif
        // prefetch next step's docWd accumulator-init tile (1 line / thread)
        if (t + 1 < DD) {
            int tn = dir ? (DD - 2 - t) : (t + 1);
            __builtin_prefetch(
                docWd + (((size_t)(bgbase + (tid & 15)) * DD + tn) << 9) + ((tid >> 4) << 4),
                0, 1);
        }

        // ---- phase 1: pre = docWd[t] (has b_g) + h @ W_r   (N=512, K=512)
        {
            v8f a0, a1;
            for (int r = 0; r < 8; ++r) {
                int mr = mhalf + r;
                size_t rowoff = ((size_t)(bgbase + mr) * DD + tt) << 9;
                a0[r] = docWd[rowoff + hc0 + n];
                a1[r] = docWd[rowoff + hc0 + 16 + n];
            }
#pragma unroll 4
            for (int kk = 0; kk < 16; ++kk) {
                const u16* bk = bpr0 + (size_t)kk * 32;
                if ((kk & 3) == 0) {
                    __builtin_prefetch(bk + 8 * 32, 0, 1);
                    __builtin_prefetch(bk + 16 * HH2 + 8 * 32, 0, 1);
                }
                v16bf af = loadA_off(aph, kk * 32);
                v16bf b0 = loadB_off(bk, 0);
                v16bf b1 = loadB_off(bk, 16 * HH2);   // 2nd N-tile: +16 rows
                a0 = __builtin_amdgcn_wmma_f32_16x16x32_bf16(
                    false, af, false, b0, (short)0, a0, false, false);
                a1 = __builtin_amdgcn_wmma_f32_16x16x32_bf16(
                    false, af, false, b1, (short)0, a1, false, false);
            }
            for (int r = 0; r < 8; ++r) {
                int mr = mhalf + r;
                presh[mr * HH2 + hc0 + n]      = a0[r];
                presh[mr * HH2 + hc0 + 16 + n] = a1[r];
            }
        }
        __syncthreads();

        // ---- phase 2: scores[b,q] = sum_h tanh(qWq + pre) * w_s + mask
        for (int pp = tid; pp < 16 * QQ; pp += 512) {
            int b = pp >> 6, q = pp & 63;
            const float4* qw4 = (const float4*)(qWq + (((size_t)(bgbase + b) * QQ + q) << 9));
            float s = 0.f;
            for (int h4 = 0; h4 < HH2 / 4; ++h4) {
                float4 v = qw4[h4];
                int h = h4 * 4;
                s += tanhf(v.x + presh[b * HH2 + h])     * wssh[h];
                s += tanhf(v.y + presh[b * HH2 + h + 1]) * wssh[h + 1];
                s += tanhf(v.z + presh[b * HH2 + h + 2]) * wssh[h + 2];
                s += tanhf(v.w + presh[b * HH2 + h + 3]) * wssh[h + 3];
            }
            float mask = (qm[(bgbase + b) * QQ + q] > 0.f) ? 0.f : -1e9f;
            scsh[pp] = s + mask;
        }
        __syncthreads();

        // ---- phase 3a: softmax over Q per batch row
        if (tid < 16) {
            float mx = -1e30f;
            for (int q = 0; q < QQ; ++q) mx = fmaxf(mx, scsh[tid * QQ + q]);
            float sum = 0.f;
            for (int q = 0; q < QQ; ++q) {
                float e = __expf(scsh[tid * QQ + q] - mx);
                scsh[tid * QQ + q] = e;
                sum += e;
            }
            float inv = 1.f / sum;
            for (int q = 0; q < QQ; ++q) scsh[tid * QQ + q] *= inv;
        }
        __syncthreads();

        // ---- phase 3b: ctx = alpha @ qry  -> ctxsh (bf16)
        {
            int b = tid >> 5;
            int cs = (tid & 31) << 4;
            float accv[16];
            for (int j = 0; j < 16; ++j) accv[j] = 0.f;
            for (int q = 0; q < QQ; ++q) {
                float a = scsh[b * QQ + q];
                const u16* qr = qrybf + (((size_t)(bgbase + b) * QQ + q) << 9) + cs;
                for (int j = 0; j < 16; ++j) accv[j] += a * bf2f(qr[j]);
            }
            for (int j = 0; j < 16; ++j)
                ctxsh[b * HH2 + cs + j] = f2bf(accv[j]);
        }
        __syncthreads();

#if defined(HAVE_TDM) || defined(HAVE_TDM_ST)
        if (tid < 32) __builtin_amdgcn_s_wait_tensorcnt(0);  // d landed, h(t-1) stored
        __syncthreads();
#endif

        // ---- phase 4: gates = [d;ctx;h] @ WgT + b_lstm   (N=2048, K=1536)
        {
            v8f acc[8];
            for (int g4 = 0; g4 < 4; ++g4)
                for (int tI = 0; tI < 2; ++tI) {
                    float bv = b_lstm[dir * NNG + g4 * HH2 + hc0 + tI * 16 + n];
                    v8f tv;
                    for (int r = 0; r < 8; ++r) tv[r] = bv;
                    acc[g4 * 2 + tI] = tv;
                }
#pragma unroll
            for (int seg = 0; seg < 3; ++seg) {
                const u16* ap = ((seg == 0) ? dsh : (seg == 1) ? ctxsh : hsh) +
                                (lane & 15) * HH2 + ((lane >> 4) << 4);
                const u16* bs = bpg0 + (size_t)(seg * 16) * 32;
#pragma unroll 4
                for (int kk2 = 0; kk2 < 16; ++kk2) {
                    const u16* bk = bs + (size_t)kk2 * 32;   // 1 add / k-step
                    v16bf af = loadA_off(ap, kk2 * 32);
#pragma unroll
                    for (int g4 = 0; g4 < 4; ++g4)
#pragma unroll
                        for (int tI = 0; tI < 2; ++tI) {
                            const int joff = (g4 * 512 + tI * 16) * KK3; // imm
                            if ((kk2 & 3) == 0)
                                __builtin_prefetch(bk + joff + 8 * 32, 0, 1);
                            v16bf bfr = loadB_off(bk, joff);
                            acc[g4 * 2 + tI] = __builtin_amdgcn_wmma_f32_16x16x32_bf16(
                                false, af, false, bfr, (short)0, acc[g4 * 2 + tI],
                                false, false);
                        }
                }
            }
            __syncthreads();   // all waves done reading the h plane

            for (int tI = 0; tI < 2; ++tI) {
                for (int r = 0; r < 8; ++r) {
                    int mr  = mhalf + r;
                    int col = hc0 + tI * 16 + n;
                    float iv = acc[0 + tI][r];
                    float fv = acc[2 + tI][r];
                    float gv = acc[4 + tI][r];
                    float ov = acc[6 + tI][r];
                    float cn = sigf(fv) * csh[mr * HH2 + col] + sigf(iv) * tanhf(gv);
                    csh[mr * HH2 + col] = cn;
                    float hn = sigf(ov) * tanhf(cn);
                    u16 hb = f2bf(hn);
                    hsh[mr * HH2 + col] = hb;
#if !defined(HAVE_TDM_ST)
                    hrbf[(((size_t)(bgbase + mr) * DD + tt) << 10) + dir * HH2 + col] = hb;
#endif
                }
            }
        }
        __syncthreads();   // hsh complete before TDM store reads it

#if defined(HAVE_TDM_ST)
        // DMA the finished h tile out; overlaps with next step's phases 1-3.
        if (tid < 32)
            tdm_store_tile(hrbf + (((size_t)bgbase * DD + tt) << 10) + dir * HH2,
                           hsh, (u32)DD * 1024u);
#endif
    }
}

// ---------------------------------------------------------------------------

extern "C" void kernel_launch(void* const* d_in, const int* in_sizes, int n_in,
                              void* d_out, int out_size, void* d_ws, size_t ws_size,
                              hipStream_t stream) {
    const float* doc    = (const float*)d_in[0];
    const float* qry    = (const float*)d_in[1];
    const float* qm     = (const float*)d_in[3];
    const float* W_q    = (const float*)d_in[4];
    const float* W_d    = (const float*)d_in[5];
    const float* W_r    = (const float*)d_in[6];
    const float* b_g    = (const float*)d_in[7];
    const float* w_s    = (const float*)d_in[8];
    const float* W_init = (const float*)d_in[9];
    const float* W_ih   = (const float*)d_in[10];
    const float* W_hh   = (const float*)d_in[11];
    const float* b_lstm = (const float*)d_in[12];
    const float* W_out  = (const float*)d_in[13];
    const float* b_out  = (const float*)d_in[14];
    float* out = (float*)d_out;
    char*  ws  = (char*)d_ws;

    u16*   docbf = (u16*)(ws + 0);               // 8,388,608 bf16
    u16*   qrybf = (u16*)(ws + 16777216);        // 1,048,576 bf16
    u16*   WqT   = (u16*)(ws + 18874368);        // 512x512
    u16*   WdT   = (u16*)(ws + 19398656);        // 512x512
    u16*   WrT   = (u16*)(ws + 19922944);        // 512x512
    u16*   WoutT = (u16*)(ws + 20447232);        // 512x1024
    u16*   WgT   = (u16*)(ws + 21495808);        // 2 x 2048x1536
    float* qWqp  = (float*)(ws + 34078720);      // 2048x512 f32
    float* docWdp= (float*)(ws + 38273024);      // 16384x512 f32
    float* qmeanp= (float*)(ws + 71827456);      // 32x512
    float* h0p   = (float*)(ws + 71892992);      // 32x512
    u16*   hrbfp = (u16*)(ws + 71958528);        // 16384x1024 bf16

    cvt_bf16_kernel<<<2048, 256, 0, stream>>>(doc, docbf, BB * DD * HH2);
    cvt_bf16_kernel<<<512, 256, 0, stream>>>(qry, qrybf, BB * QQ * HH2);

    transpose_bf16_kernel<<<512, 256, 0, stream>>>(W_q, WqT, HH2, HH2);
    transpose_bf16_kernel<<<512, 256, 0, stream>>>(W_d, WdT, HH2, HH2);
    transpose_bf16_kernel<<<512, 256, 0, stream>>>(W_r, WrT, HH2, HH2);
    transpose_bf16_kernel<<<1024, 256, 0, stream>>>(W_out, WoutT, 2 * HH2, HH2);
    build_wgt_kernel<<<2048, 256, 0, stream>>>(W_ih, W_hh, WgT);
    build_wgt_kernel<<<2048, 256, 0, stream>>>(W_ih + (size_t)1024 * NNG,
                                               W_hh + (size_t)512 * NNG,
                                               WgT + (size_t)NNG * KK3);

    qmean_kernel<<<64, 256, 0, stream>>>(qry, qm, qmeanp);
    h0_kernel<<<64, 256, 0, stream>>>(qmeanp, W_init, h0p);

    gemm_bf16_tn<512><<<dim3(BB * QQ / 16, 1), 256, 0, stream>>>(
        qrybf, WqT, nullptr, qWqp, BB * QQ, HH2);
    gemm_bf16_tn<512><<<dim3(BB * DD / 16, 1), 256, 0, stream>>>(
        docbf, WdT, b_g, docWdp, BB * DD, HH2);

    matchlstm_rnn_kernel<<<4, 512, 120832, stream>>>(
        docbf, docWdp, qWqp, qrybf, qm, w_s, b_lstm, WrT, WgT, h0p, hrbfp);

    gemm_bf16_tn<1024><<<dim3(BB * DD / 16, 1), 256, 0, stream>>>(
        hrbfp, WoutT, b_out, out, BB * DD, HH2);
}
